// STRNNM_74783970558553
// MI455X (gfx1250) — matrix operand order repaired and checked by reading, not compiled, not run on hardware
//
#include <hip/hip_runtime.h>
#include <hip/hip_fp16.h>
#include <math.h>

typedef _Float16 f16;
typedef _Float16 v8h  __attribute__((ext_vector_type(8)));
typedef _Float16 v16h __attribute__((ext_vector_type(16)));
typedef float    v8f  __attribute__((ext_vector_type(8)));

// Problem constants
constexpr int Bc  = 16;
constexpr int Sc  = 200;
constexpr int INc = 128;
constexpr int Hc  = 128;
constexpr int Kc  = 10;
constexpr int OUTc = 10000;
constexpr int Mc  = Bc * Sc;        // 3200 token rows
constexpr int NSc = Kc + 1;         // 11 bucket slots
constexpr int ZSc = 2 * NSc;        // 22 (time slots then dist slots)
constexpr int FHc = 4 * Hc;         // 512
constexpr float TWc = 1.0f, DWc = 1.0f;

// ---- workspace layout (bytes, all 256-aligned) ----
constexpr size_t OFF_XA   = 0;                                  // f16 [Mc][INc]
constexpr size_t SZ_XA    = (size_t)Mc * INc * 2;               // 819200
constexpr size_t OFF_WTBT = OFF_XA + SZ_XA;                     // f16 [ZSc][Hc][INc] (B transposed)
constexpr size_t SZ_WTBT  = (size_t)ZSc * Hc * INc * 2;         // 720896
constexpr size_t OFF_PROJ = OFF_WTBT + SZ_WTBT;                 // f16 [ZSc][Mc][Hc]
constexpr size_t SZ_PROJ  = (size_t)ZSc * Mc * Hc * 2;          // 18022400
constexpr size_t OFF_X    = OFF_PROJ + SZ_PROJ;                 // f32 [Bc][Sc][Hc]
constexpr size_t SZ_X     = (size_t)Mc * Hc * 4;                // 1638400
constexpr size_t OFF_HPRE = OFF_X + SZ_X;                       // f32 [Bc][Hc]
constexpr size_t SZ_HPRE  = (size_t)Bc * Hc * 4;                // 8192
constexpr size_t OFF_TA   = OFF_HPRE + SZ_HPRE;                 // f16 [Bc][FHc]
constexpr size_t SZ_TA    = (size_t)Bc * FHc * 2;               // 16384
constexpr size_t OFF_W2T  = OFF_TA + SZ_TA;                     // f16 [OUTc][FHc] (W2 transposed)
constexpr size_t SZ_W2T   = (size_t)OUTc * FHc * 2;             // 10240000

// ---------------------------------------------------------------------------
// WMMA fragment loaders, matching CDNA5 ISA §7.12.2 VGPR striping (wave32).
// A (16x32 f16): lane L holds row L%16; K-octets {0..7,16..23} for L<16,
//                {8..15,24..31} for L>=16  -> two contiguous 16B loads.
// B (32x16 f16), stored transposed (N-major): lane L holds col L%16,
//                K-octets {0..7,16..23} for L<16, {8..15,24..31} for L>=16.
// ---------------------------------------------------------------------------
__device__ inline v16h load_a_frag(const f16* __restrict__ A, int lda,
                                   int row0, int kb, int lane) {
    const int r  = row0 + (lane & 15);
    const int k0 = kb * 32 + ((lane < 16) ? 0 : 8);
    const f16* p = A + (size_t)r * lda + k0;
    v16h a;
    *(v8h*)&a       = *(const v8h*)p;
    *((v8h*)&a + 1) = *(const v8h*)(p + 16);
    return a;
}

__device__ inline v16h load_b_frag(const f16* __restrict__ BT, int ldb,
                                   int col0, int kb, int lane) {
    const int n  = col0 + (lane & 15);
    const int k0 = kb * 32 + ((lane < 16) ? 0 : 8);
    const f16* p = BT + (size_t)n * ldb + k0;
    v16h b;
    *(v8h*)&b       = *(const v8h*)p;
    *((v8h*)&b + 1) = *(const v8h*)(p + 16);
    return b;
}

// ---------------------------------------------------------------------------
// Prep kernels: gather + fp32 -> f16 conversion / transposition
// ---------------------------------------------------------------------------
__global__ void gather_embed(const int* __restrict__ seq,
                             const float* __restrict__ tab,
                             f16* __restrict__ xA) {
    int idx = blockIdx.x * blockDim.x + threadIdx.x;     // Mc*INc
    if (idx >= Mc * INc) return;
    int m = idx >> 7, h = idx & 127;
    xA[idx] = (f16)tab[(size_t)seq[m] * INc + h];
}

__global__ void prep_weights(const float* __restrict__ tw,
                             const float* __restrict__ dw,
                             f16* __restrict__ bt) {
    int idx = blockIdx.x * blockDim.x + threadIdx.x;     // ZSc*Hc*INc
    if (idx >= ZSc * Hc * INc) return;
    int z   = idx / (Hc * INc);
    int rem = idx % (Hc * INc);
    int o   = rem / INc;
    int ii  = rem % INc;
    const float* src = (z < NSc) ? tw : dw;
    int slot = (z < NSc) ? z : z - NSc;
    // source layout [slot][i][o]; dest [z][o][i]
    bt[idx] = (f16)src[((size_t)slot * INc + ii) * Hc + o];
}

__global__ void prep_w2(const float* __restrict__ W2, f16* __restrict__ w2t) {
    int idx = blockIdx.x * blockDim.x + threadIdx.x;     // OUTc*FHc
    if (idx >= OUTc * FHc) return;
    int o = idx / FHc, k = idx % FHc;
    w2t[idx] = (f16)W2[(size_t)k * OUTc + o];            // [k][o] -> [o][k]
}

// ---------------------------------------------------------------------------
// Projection GEMM: proj[z] = xA[3200x128] @ W[z][128x128], f32 acc -> f16 out
// one wave per 16x16 tile, K-loop of 4 x (16x16x32 WMMA)
// ---------------------------------------------------------------------------
__global__ void proj_gemm(const f16* __restrict__ xA,
                          const f16* __restrict__ wtBT,
                          f16* __restrict__ proj) {
    const int lane = threadIdx.x & 31;
    const int mt = blockIdx.x;           // 0..199
    const int nt = blockIdx.y;           // 0..7
    const int z  = blockIdx.z;           // 0..21
    const f16* BT = wtBT + (size_t)z * Hc * INc;
    f16* C        = proj + (size_t)z * Mc * Hc;
    v8f c = {};
#pragma unroll
    for (int kb = 0; kb < INc / 32; ++kb) {
        v16h a = load_a_frag(xA, INc, mt * 16, kb, lane);
        v16h b = load_b_frag(BT, INc, nt * 16, kb, lane);
        c = __builtin_amdgcn_wmma_f32_16x16x32_f16(false, a, false, b,
                                                   (short)0, c, false, false);
    }
    const int n  = nt * 16 + (lane & 15);
    const int m0 = mt * 16 + ((lane < 16) ? 0 : 8);
#pragma unroll
    for (int v = 0; v < 8; ++v)
        C[(size_t)(m0 + v) * Hc + n] = (f16)c[v];
}

// ---------------------------------------------------------------------------
// Masked gather-sum over context j: X[b,i,:] (the memory-bound hot loop)
// one 128-thread block per (b,i); timestamp/lat/lng staged in LDS
// ---------------------------------------------------------------------------
__global__ void gather_sum(const float* __restrict__ ts,
                           const float* __restrict__ lat,
                           const float* __restrict__ lng,
                           const int* __restrict__ length,
                           const f16* __restrict__ proj,
                           float* __restrict__ X) {
    __shared__ float sts[Sc], sla[Sc], sln[Sc];
    const int bi = blockIdx.x;           // 0..3199
    const int b = bi / Sc, i = bi % Sc;
    const int h = threadIdx.x;           // 0..127
    for (int j = h; j < Sc; j += 128) {
        sts[j] = ts[b * Sc + j];
        sla[j] = lat[b * Sc + j];
        sln[j] = lng[b * Sc + j];
    }
    __syncthreads();
    const int len = length[b];
    const float ti = sts[i], lai = sla[i], lni = sln[i];
    float acc = 0.0f;
    for (int j = 0; j <= i; ++j) {
        if (j >= len) continue;                       // valid_j: j < length[b]
        float td = ti - sts[j];
        if (td < 0.0f || td > TWc) continue;          // time window
        float dx = lai - sla[j], dy = lni - sln[j];
        float dist = sqrtf(dx * dx + dy * dy);
        int tix = (int)floorf(td * (float)Kc / TWc);              if (tix > Kc) tix = Kc;
        int dix = (int)floorf(fminf(dist, DWc) * (float)Kc / DWc); if (dix > Kc) dix = Kc;
        size_t mj = (size_t)(b * Sc + j);
        float pt = (float)proj[((size_t)tix * Mc + mj) * Hc + h];
        float pd = (float)proj[((size_t)(NSc + dix) * Mc + mj) * Hc + h];
        acc += 0.5f * (pt + pd);
    }
    X[(size_t)bi * Hc + h] = acc;
}

// ---------------------------------------------------------------------------
// Sequential sigmoid recurrence, single persistent block (latency bound);
// captures h at step length[b]-1 into h_pre.
// ---------------------------------------------------------------------------
__global__ void __launch_bounds__(1024)
rnn_scan(const float* __restrict__ X, const float* __restrict__ Wh,
         const int* __restrict__ length, float* __restrict__ h_pre) {
    __shared__ float hs[Bc * Hc];        // 2048 floats
    const int tid = threadIdx.x;         // 0..1023, 2 elements each
    for (int e = tid; e < Bc * Hc; e += 1024) hs[e] = 0.0f;
    __syncthreads();
    for (int i = 0; i < Sc; ++i) {
        float a[2];
#pragma unroll
        for (int p = 0; p < 2; ++p) {
            int e = tid + p * 1024;
            int b = e >> 7, n = e & 127;
            float acc = X[((size_t)b * Sc + i) * Hc + n];
            for (int k = 0; k < Hc; ++k)
                acc += hs[b * Hc + k] * Wh[k * Hc + n];
            a[p] = 1.0f / (1.0f + __expf(-acc));
        }
        __syncthreads();
#pragma unroll
        for (int p = 0; p < 2; ++p) hs[tid + p * 1024] = a[p];
        __syncthreads();
#pragma unroll
        for (int p = 0; p < 2; ++p) {
            int e = tid + p * 1024;
            int b = e >> 7;
            if (length[b] - 1 == i) h_pre[e] = a[p];
        }
    }
}

// ---------------------------------------------------------------------------
// t = tanh(h_pre @ W1 + b1), converted to f16 A-matrix for the W2 WMMA GEMM
// ---------------------------------------------------------------------------
__global__ void mlp1(const float* __restrict__ h_pre, const float* __restrict__ W1,
                     const float* __restrict__ b1, f16* __restrict__ tA) {
    int idx = blockIdx.x * blockDim.x + threadIdx.x;   // Bc*FHc = 8192
    if (idx >= Bc * FHc) return;
    int b = idx / FHc, n = idx % FHc;
    float acc = b1[n];
    for (int k = 0; k < Hc; ++k) acc += h_pre[b * Hc + k] * W1[k * FHc + n];
    tA[idx] = (f16)tanhf(acc);
}

// ---------------------------------------------------------------------------
// Output GEMM: out[16x10000] = tA[16x512] @ W2 + b2, via WMMA f16, f32 acc.
// M fits in a single tile row; one wave per 16-wide N tile, K-loop of 16.
// ---------------------------------------------------------------------------
__global__ void out_gemm(const f16* __restrict__ tA, const f16* __restrict__ W2T,
                         const float* __restrict__ b2, float* __restrict__ out) {
    const int lane = threadIdx.x & 31;
    const int nt = blockIdx.x;           // 0..624
    v8f c = {};
#pragma unroll
    for (int kb = 0; kb < FHc / 32; ++kb) {
        v16h a = load_a_frag(tA, FHc, 0, kb, lane);
        v16h b = load_b_frag(W2T, FHc, nt * 16, kb, lane);
        c = __builtin_amdgcn_wmma_f32_16x16x32_f16(false, a, false, b,
                                                   (short)0, c, false, false);
    }
    const int n  = nt * 16 + (lane & 15);
    const int r0 = (lane < 16) ? 0 : 8;
    const float bias = b2[n];
#pragma unroll
    for (int v = 0; v < 8; ++v)
        out[(size_t)(r0 + v) * OUTc + n] = c[v] + bias;
}

// ---------------------------------------------------------------------------
extern "C" void kernel_launch(void* const* d_in, const int* in_sizes, int n_in,
                              void* d_out, int out_size, void* d_ws, size_t ws_size,
                              hipStream_t stream) {
    (void)in_sizes; (void)n_in; (void)out_size; (void)ws_size;

    const int*   seq = (const int*)d_in[0];
    const int*   len = (const int*)d_in[1];
    const float* ts  = (const float*)d_in[2];
    const float* lat = (const float*)d_in[3];
    const float* lng = (const float*)d_in[4];
    const float* tab = (const float*)d_in[5];
    const float* tw  = (const float*)d_in[6];
    const float* dw  = (const float*)d_in[7];
    const float* Wh  = (const float*)d_in[8];
    const float* W1  = (const float*)d_in[9];
    const float* b1  = (const float*)d_in[10];
    const float* W2  = (const float*)d_in[11];
    const float* b2  = (const float*)d_in[12];
    float* out = (float*)d_out;

    char* ws = (char*)d_ws;
    f16*   xA    = (f16*)(ws + OFF_XA);
    f16*   wtBT  = (f16*)(ws + OFF_WTBT);
    f16*   proj  = (f16*)(ws + OFF_PROJ);
    float* X     = (float*)(ws + OFF_X);
    float* h_pre = (float*)(ws + OFF_HPRE);
    f16*   tA    = (f16*)(ws + OFF_TA);
    f16*   W2T   = (f16*)(ws + OFF_W2T);

    gather_embed<<<(Mc * INc + 255) / 256, 256, 0, stream>>>(seq, tab, xA);
    prep_weights<<<(ZSc * Hc * INc + 255) / 256, 256, 0, stream>>>(tw, dw, wtBT);
    prep_w2<<<(OUTc * FHc + 255) / 256, 256, 0, stream>>>(W2, W2T);

    dim3 g1(Mc / 16, Hc / 16, ZSc);      // 200 x 8 x 22 tiles, 1 wave each
    proj_gemm<<<g1, 32, 0, stream>>>(xA, wtBT, proj);

    gather_sum<<<Mc, 128, 0, stream>>>(ts, lat, lng, len, proj, X);

    rnn_scan<<<1, 1024, 0, stream>>>(X, Wh, len, h_pre);

    mlp1<<<(Bc * FHc + 255) / 256, 256, 0, stream>>>(h_pre, W1, b1, tA);

    out_gemm<<<OUTc / 16, 32, 0, stream>>>(tA, W2T, b2, out);
}
